// BiLSTM_CRF_BERT_83665962926636
// MI455X (gfx1250) — matrix operand order repaired and checked by reading, not compiled, not run on hardware
//
#include <hip/hip_runtime.h>
#include <math.h>

// ---------------- types / helpers ----------------
typedef __attribute__((ext_vector_type(16))) __bf16 bf16x16;
typedef __attribute__((ext_vector_type(8)))  float  f32x8;

union Frag16 { bf16x16 v; uint4 q[2]; };

// native gfx1250 bf16 converts (single v_cvt op, RTNE in hardware)
static __device__ inline __bf16 f2bf(float f) { return (__bf16)f; }
static __device__ inline float bf2f(__bf16 b) { return (float)b; }
static __device__ inline __bf16 bfzero() { return (__bf16)0.0f; }

static __device__ inline f32x8 wmma_bf16(bf16x16 a, bf16x16 b, f32x8 c) {
  return __builtin_amdgcn_wmma_f32_16x16x32_bf16(false, a, false, b, (short)0, c, false, false);
}
static __device__ inline float sigf(float x) { return 1.0f / (1.0f + expf(-x)); }

#define CB 32
#define CS 75
#define CH 768
#define CL 12
#define CNH 12
#define CDH 64
#define CFF 3072
#define CHID 256
#define CLH 128
#define CT 17
#define CM (CB*CS)          // 2400 rows
#define SPAD 80             // padded S for attention tiles
#define KPAD 96             // padded K for probs*V
#define GY ((CM + 63) / 64) // 38 M-blocks of 64 rows

// ---------------- embedding + layernorm ----------------
__global__ __launch_bounds__(256)
void embed_ln_kernel(const int* ids, const float* we, const float* pe,
                     const float* te, const float* g, const float* b,
                     float* xo, __bf16* xbo) {
  int tok = blockIdx.x;            // 0..2399
  int s = tok % CS;
  int id = ids[tok];
  __shared__ float red[256];
  float v[3];
  float part = 0.f;
  #pragma unroll
  for (int j = 0; j < 3; j++) {
    int i = threadIdx.x + j * 256;
    v[j] = we[(size_t)id * CH + i] + pe[(size_t)s * CH + i] + te[i];
    part += v[j];
  }
  red[threadIdx.x] = part; __syncthreads();
  for (int o = 128; o; o >>= 1) { if (threadIdx.x < o) red[threadIdx.x] += red[threadIdx.x + o]; __syncthreads(); }
  float mean = red[0] / (float)CH; __syncthreads();
  part = 0.f;
  #pragma unroll
  for (int j = 0; j < 3; j++) { float d = v[j] - mean; part += d * d; }
  red[threadIdx.x] = part; __syncthreads();
  for (int o = 128; o; o >>= 1) { if (threadIdx.x < o) red[threadIdx.x] += red[threadIdx.x + o]; __syncthreads(); }
  float rstd = rsqrtf(red[0] / (float)CH + 1e-12f);
  #pragma unroll
  for (int j = 0; j < 3; j++) {
    int i = threadIdx.x + j * 256;
    float y = (v[j] - mean) * rstd * g[i] + b[i];
    xo[(size_t)tok * CH + i] = y;
    xbo[(size_t)tok * CH + i] = f2bf(y);
  }
}

// ---------------- residual add + layernorm ----------------
__global__ __launch_bounds__(256)
void addln_kernel(const float* x, const float* a, const float* g, const float* b,
                  float* xo, __bf16* xbo) {
  int tok = blockIdx.x;
  __shared__ float red[256];
  float v[3]; float part = 0.f;
  #pragma unroll
  for (int j = 0; j < 3; j++) {
    int i = threadIdx.x + j * 256;
    v[j] = x[(size_t)tok * CH + i] + a[(size_t)tok * CH + i];
    part += v[j];
  }
  red[threadIdx.x] = part; __syncthreads();
  for (int o = 128; o; o >>= 1) { if (threadIdx.x < o) red[threadIdx.x] += red[threadIdx.x + o]; __syncthreads(); }
  float mean = red[0] / (float)CH; __syncthreads();
  part = 0.f;
  #pragma unroll
  for (int j = 0; j < 3; j++) { float d = v[j] - mean; part += d * d; }
  red[threadIdx.x] = part; __syncthreads();
  for (int o = 128; o; o >>= 1) { if (threadIdx.x < o) red[threadIdx.x] += red[threadIdx.x + o]; __syncthreads(); }
  float rstd = rsqrtf(red[0] / (float)CH + 1e-12f);
  #pragma unroll
  for (int j = 0; j < 3; j++) {
    int i = threadIdx.x + j * 256;
    float y = (v[j] - mean) * rstd * g[i] + b[i];
    xo[(size_t)tok * CH + i] = y;
    xbo[(size_t)tok * CH + i] = f2bf(y);
  }
}

// ---------------- generic bf16(A) x fp32(B, converted on stage) GEMM ----------------
// C[M,N] = A[M,K](bf16) * B[K,N](fp32) + bias ; act 0=none 1=gelu(exact)
// block: 256 thr = 8 waves; block tile 64(M) x 128(N).
// wave w owns N-subtile w (16 cols) and iterates 4 M-subtiles reusing one B fragment.
__global__ __launch_bounds__(256)
void gemm_kernel(const __bf16* __restrict__ A, int lda,
                 const float* __restrict__ Bw, int ldb,
                 const float* __restrict__ bias,
                 float* Cf, __bf16* Cb, int ldc,
                 int M, int N, int K, int act) {
  __shared__ __bf16 Bs[128][32];     // [col][k], contiguous along k
  int m0 = blockIdx.y * 64;
  int n0 = blockIdx.x * 128;
  int lane = threadIdx.x & 31;
  int wave = threadIdx.x >> 5;
  f32x8 acc[4] = {{}, {}, {}, {}};
  int ac = (lane >> 4) * 8;
  int bk = (lane >> 4) * 8;
  int bcol = wave * 16 + (lane & 15);
  const __bf16* Arow[4];
  #pragma unroll
  for (int i = 0; i < 4; i++) {
    int r = m0 + i * 16 + (lane & 15);
    if (r > M - 1) r = M - 1;                 // clamp: keeps EXEC all-ones for WMMA
    Arow[i] = A + (size_t)r * lda;
  }
  const bool fullN = (n0 + 128 <= N);         // uniform across block
  // hoisted per-thread staging pointers (fast path): advance by 32*ldb per k-step
  const float* gptr[4];
  int sn4[4];
  #pragma unroll
  for (int j = 0; j < 4; j++) {
    int idx4 = j * 256 + threadIdx.x;         // 0..1023
    int kk = idx4 >> 5;                       // 0..31
    sn4[j] = idx4 & 31;
    gptr[j] = Bw + (size_t)kk * ldb + n0 + sn4[j] * 4;
  }
  const size_t ldstep = (size_t)32 * ldb;
  for (int k0 = 0; k0 < K; k0 += 32) {
    if (fullN) {
      if (k0 + 32 < K) __builtin_prefetch(gptr[0] + ldstep, 0, 1);
      #pragma unroll
      for (int j = 0; j < 4; j++) {
        int idx4 = j * 256 + threadIdx.x;
        int kk = idx4 >> 5;
        float4 vv = *(const float4*)gptr[j];
        gptr[j] += ldstep;
        Bs[sn4[j] * 4 + 0][kk] = f2bf(vv.x);
        Bs[sn4[j] * 4 + 1][kk] = f2bf(vv.y);
        Bs[sn4[j] * 4 + 2][kk] = f2bf(vv.z);
        Bs[sn4[j] * 4 + 3][kk] = f2bf(vv.w);
      }
    } else {
      #pragma unroll
      for (int j = 0; j < 16; j++) {
        int idx = j * 256 + threadIdx.x;      // 0..4095
        int kk = idx >> 7;
        int nn = idx & 127;
        int gn = n0 + nn;
        float val = (gn < N) ? Bw[(size_t)(k0 + kk) * ldb + gn] : 0.0f;
        Bs[nn][kk] = f2bf(val);
      }
    }
    __syncthreads();
    Frag16 fb;
    fb.q[0] = *(const uint4*)(&Bs[bcol][bk]);
    fb.q[1] = *(const uint4*)(&Bs[bcol][bk + 16]);
    #pragma unroll
    for (int i = 0; i < 4; i++) {
      Frag16 fa;
      fa.q[0] = *(const uint4*)(Arow[i] + k0 + ac);
      fa.q[1] = *(const uint4*)(Arow[i] + k0 + ac + 16);
      acc[i] = wmma_bf16(fa.v, fb.v, acc[i]);
    }
    __syncthreads();
  }
  int col = n0 + wave * 16 + (lane & 15);
  float bi = (bias && col < N) ? bias[col] : 0.0f;
  #pragma unroll
  for (int i = 0; i < 4; i++) {
    int rb = m0 + i * 16 + (lane >> 4) * 8;
    #pragma unroll
    for (int r = 0; r < 8; r++) {
      int row = rb + r;
      if (row < M && col < N) {
        float y = acc[i][r] + bi;
        if (act == 1) y = 0.5f * y * (1.0f + erff(y * 0.70710678118f));
        if (Cf) Cf[(size_t)row * ldc + col] = y;
        if (Cb) Cb[(size_t)row * ldc + col] = f2bf(y);
      }
    }
  }
}

// ---------------- attention scores: Sc = Q K^T / 8 + mask bias ----------------
__global__ __launch_bounds__(256)
void scores_kernel(const __bf16* __restrict__ q, const __bf16* __restrict__ k,
                   const int* __restrict__ mask, float* __restrict__ sc) {
  int bh = blockIdx.x;
  int b = bh / CNH, h = bh % CNH;
  int lane = threadIdx.x & 31;
  int wave = threadIdx.x >> 5;
  int ac = (lane >> 4) * 8;
  int bk = (lane >> 4) * 8;
  for (int ti = wave; ti < 25; ti += 8) {
    int mt = ti / 5, nt = ti % 5;
    int srow = mt * 16 + (lane & 15); if (srow > CS - 1) srow = CS - 1;
    int scol = nt * 16 + (lane & 15); if (scol > CS - 1) scol = CS - 1;
    const __bf16* Aq = q + ((size_t)(b * CS + srow)) * CH + h * CDH;
    const __bf16* Bk = k + ((size_t)(b * CS + scol)) * CH + h * CDH;
    f32x8 acc = {};
    #pragma unroll
    for (int k0 = 0; k0 < CDH; k0 += 32) {
      Frag16 fa, fb;
      fa.q[0] = *(const uint4*)(Aq + k0 + ac);
      fa.q[1] = *(const uint4*)(Aq + k0 + ac + 16);
      fb.q[0] = *(const uint4*)(Bk + k0 + bk);
      fb.q[1] = *(const uint4*)(Bk + k0 + bk + 16);
      acc = wmma_bf16(fa.v, fb.v, acc);
    }
    int col = nt * 16 + (lane & 15);
    int rb = mt * 16 + (lane >> 4) * 8;
    float mb = 0.0f;
    if (col < CS) mb = (1.0f - (float)mask[b * CS + col]) * -10000.0f;
    #pragma unroll
    for (int r = 0; r < 8; r++) {
      int row = rb + r;
      if (row < CS && col < CS)
        sc[((size_t)bh * CS + row) * CS + col] = acc[r] * 0.125f + mb;
    }
  }
}

// ---------------- softmax over key dim, writes zero-padded bf16 probs ----------------
__global__ __launch_bounds__(128)
void softmax_kernel(const float* __restrict__ sc, __bf16* __restrict__ pr) {
  int row = blockIdx.x * 4 + (threadIdx.x >> 5);   // 0..28799
  int lane = threadIdx.x & 31;
  int bh = row / CS, qr = row % CS;
  const float* r = sc + (size_t)row * CS;
  float v0 = (lane      < CS) ? r[lane]      : -3.0e38f;
  float v1 = (lane + 32 < CS) ? r[lane + 32] : -3.0e38f;
  float v2 = (lane + 64 < CS) ? r[lane + 64] : -3.0e38f;
  float m = fmaxf(v0, fmaxf(v1, v2));
  for (int o = 16; o; o >>= 1) m = fmaxf(m, __shfl_xor(m, o, 32));
  float e0 = (lane      < CS) ? expf(v0 - m) : 0.f;
  float e1 = (lane + 32 < CS) ? expf(v1 - m) : 0.f;
  float e2 = (lane + 64 < CS) ? expf(v2 - m) : 0.f;
  float s = e0 + e1 + e2;
  for (int o = 16; o; o >>= 1) s += __shfl_xor(s, o, 32);
  float inv = 1.0f / s;
  __bf16* ow = pr + ((size_t)bh * SPAD + qr) * KPAD;
  if (lane      < CS) ow[lane]      = f2bf(e0 * inv);
  if (lane + 32 < CS) ow[lane + 32] = f2bf(e1 * inv);
  if (lane + 64 < CS) ow[lane + 64] = f2bf(e2 * inv);
}

// ---------------- ctx = P(80x96, zero padded) * V(96x64) ----------------
__global__ __launch_bounds__(256)
void ctx_kernel(const __bf16* __restrict__ pr, const __bf16* __restrict__ v,
                __bf16* __restrict__ ctx) {
  __shared__ __bf16 Vs[CDH][KPAD];   // [col][k]
  int bh = blockIdx.x;
  int b = bh / CNH, h = bh % CNH;
  for (int i = threadIdx.x; i < CDH * KPAD; i += 256) {
    int c = i / KPAD, kk = i % KPAD;
    __bf16 val = bfzero();
    if (kk < CS) val = v[((size_t)(b * CS + kk)) * CH + h * CDH + c];
    Vs[c][kk] = val;
  }
  __syncthreads();
  int lane = threadIdx.x & 31;
  int wave = threadIdx.x >> 5;
  int ac = (lane >> 4) * 8;
  int bk = (lane >> 4) * 8;
  const __bf16* Ab = pr + (size_t)bh * SPAD * KPAD;
  for (int ti = wave; ti < 20; ti += 8) {    // 5 row tiles x 4 col tiles
    int mt = ti / 4, nt = ti % 4;
    const __bf16* Ar = Ab + (size_t)(mt * 16 + (lane & 15)) * KPAD;
    int bcol = nt * 16 + (lane & 15);
    f32x8 acc = {};
    #pragma unroll
    for (int k0 = 0; k0 < KPAD; k0 += 32) {
      Frag16 fa, fb;
      fa.q[0] = *(const uint4*)(Ar + k0 + ac);
      fa.q[1] = *(const uint4*)(Ar + k0 + ac + 16);
      fb.q[0] = *(const uint4*)(&Vs[bcol][k0 + bk]);
      fb.q[1] = *(const uint4*)(&Vs[bcol][k0 + bk + 16]);
      acc = wmma_bf16(fa.v, fb.v, acc);
    }
    int col = h * CDH + nt * 16 + (lane & 15);
    int rb = mt * 16 + (lane >> 4) * 8;
    #pragma unroll
    for (int r = 0; r < 8; r++) {
      int row = rb + r;
      if (row < CS) ctx[((size_t)(b * CS + row)) * CH + col] = f2bf(acc[r]);
    }
  }
}

// ---------------- small utility kernels ----------------
__global__ void transpose_f32_kernel(const float* in, float* out, int rows, int cols) {
  int idx = blockIdx.x * 256 + threadIdx.x;
  if (idx < rows * cols) {
    int r = idx / cols, c = idx % cols;
    out[(size_t)c * rows + r] = in[(size_t)r * cols + c];
  }
}
__global__ void cvt_bf16_kernel(const float* in, __bf16* out, int n) {
  int idx = blockIdx.x * 256 + threadIdx.x;
  if (idx < n) out[idx] = f2bf(in[idx]);
}
__global__ void addvec_kernel(const float* a, const float* b, float* o, int n) {
  int idx = blockIdx.x * 256 + threadIdx.x;
  if (idx < n) o[idx] = a[idx] + b[idx];
}

// ---------------- persistent BiLSTM (one block per direction) ----------------
__global__ __launch_bounds__(512)
void lstm_kernel(const float* __restrict__ xp, const __bf16* __restrict__ WhhT,
                 __bf16* __restrict__ out) {
  int dir = blockIdx.x;
  __shared__ __bf16 hb[CB][CLH];       // 8 KB
  __shared__ float  cb[CB][CLH];       // 16 KB
  __shared__ __bf16 Gb[CB][4 * CLH];   // 32 KB
  for (int i = threadIdx.x; i < CB * CLH; i += 512) {
    (&hb[0][0])[i] = bfzero();
    (&cb[0][0])[i] = 0.0f;
  }
  __syncthreads();
  const float* xpd = xp + (size_t)dir * CM * 4 * CLH;
  const __bf16* Wd = WhhT + (size_t)dir * 4 * CLH * CLH;
  int lane = threadIdx.x & 31;
  int wave = threadIdx.x >> 5;
  int ac = (lane >> 4) * 8;
  int bk = (lane >> 4) * 8;
  for (int t = 0; t < CS; t++) {
    int tt = dir ? (CS - 1 - t) : t;
    #pragma unroll
    for (int j = 0; j < 4; j++) {
      int ti = wave * 4 + j;           // 0..63
      int mt = ti & 1, nt = ti >> 1;
      int ar = mt * 16 + (lane & 15);
      int bc = nt * 16 + (lane & 15);
      const __bf16* bp = Wd + (size_t)bc * CLH;
      f32x8 acc = {};
      #pragma unroll
      for (int k0 = 0; k0 < CLH; k0 += 32) {
        Frag16 fa, fb;
        fa.q[0] = *(const uint4*)(&hb[ar][k0 + ac]);
        fa.q[1] = *(const uint4*)(&hb[ar][k0 + ac + 16]);
        fb.q[0] = *(const uint4*)(bp + k0 + bk);
        fb.q[1] = *(const uint4*)(bp + k0 + bk + 16);
        acc = wmma_bf16(fa.v, fb.v, acc);
      }
      int col = nt * 16 + (lane & 15);
      int rb = mt * 16 + (lane >> 4) * 8;
      #pragma unroll
      for (int r = 0; r < 8; r++) {
        int bb = rb + r;
        float g = acc[r] + xpd[((size_t)bb * CS + tt) * (4 * CLH) + col];
        Gb[bb][col] = f2bf(g);
      }
    }
    __syncthreads();
    #pragma unroll
    for (int j = 0; j < 8; j++) {
      int idx = threadIdx.x + j * 512;     // 0..4095
      int bb = idx >> 7, jj = idx & 127;
      float i_ = sigf(bf2f(Gb[bb][jj]));
      float f_ = sigf(bf2f(Gb[bb][CLH + jj]));
      float g_ = tanhf(bf2f(Gb[bb][2 * CLH + jj]));
      float o_ = sigf(bf2f(Gb[bb][3 * CLH + jj]));
      float c = f_ * cb[bb][jj] + i_ * g_;
      float h = o_ * tanhf(c);
      cb[bb][jj] = c;
      __bf16 hh = f2bf(h);
      hb[bb][jj] = hh;
      out[((size_t)bb * CS + tt) * CHID + dir * CLH + jj] = hh;
    }
    __syncthreads();
  }
}

// ---------------- CRF forward (negative log likelihood, sum-reduced) ----------------
__global__ void crf_nll_kernel(const float* __restrict__ em, const int* __restrict__ tags,
                               const float* start, const float* endv, const float* trans,
                               float* out) {
  __shared__ float tr[CT * CT], st[CT], en[CT];
  for (int i = threadIdx.x; i < CT * CT; i += 32) tr[i] = trans[i];
  for (int i = threadIdx.x; i < CT; i += 32) { st[i] = start[i]; en[i] = endv[i]; }
  __syncthreads();
  int b = threadIdx.x;
  const float* eb = em + (size_t)b * CS * CT;
  const int* tg = tags + b * CS;
  float num = st[tg[0]] + en[tg[CS - 1]];
  for (int s = 0; s < CS; s++) num += eb[s * CT + tg[s]];
  for (int s = 0; s < CS - 1; s++) num += tr[tg[s] * CT + tg[s + 1]];
  float alpha[CT], na[CT];
  for (int j = 0; j < CT; j++) alpha[j] = st[j] + eb[j];
  for (int s = 1; s < CS; s++) {
    for (int j = 0; j < CT; j++) {
      float m = -3.0e38f;
      for (int i = 0; i < CT; i++) m = fmaxf(m, alpha[i] + tr[i * CT + j]);
      float sum = 0.f;
      for (int i = 0; i < CT; i++) sum += expf(alpha[i] + tr[i * CT + j] - m);
      na[j] = m + logf(sum) + eb[s * CT + j];
    }
    for (int j = 0; j < CT; j++) alpha[j] = na[j];
  }
  float m = -3.0e38f;
  for (int j = 0; j < CT; j++) m = fmaxf(m, alpha[j] + en[j]);
  float sum = 0.f;
  for (int j = 0; j < CT; j++) sum += expf(alpha[j] + en[j] - m);
  float part = num - (m + logf(sum));
  for (int o = 16; o; o >>= 1) part += __shfl_down(part, o, 32);
  if (b == 0) out[0] = -part;
}

// ---------------- Viterbi decode ----------------
__global__ void viterbi_kernel(const float* __restrict__ em,
                               const float* start, const float* endv, const float* trans,
                               float* out) {
  __shared__ unsigned char bp[CB][CS - 1][CT];   // ~40 KB
  __shared__ float tr[CT * CT], st[CT], en[CT];
  for (int i = threadIdx.x; i < CT * CT; i += 32) tr[i] = trans[i];
  for (int i = threadIdx.x; i < CT; i += 32) { st[i] = start[i]; en[i] = endv[i]; }
  __syncthreads();
  int b = threadIdx.x;
  const float* eb = em + (size_t)b * CS * CT;
  float alpha[CT], na[CT];
  for (int j = 0; j < CT; j++) alpha[j] = st[j] + eb[j];
  for (int s = 1; s < CS; s++) {
    for (int j = 0; j < CT; j++) {
      float best = -3.0e38f; int bi = 0;
      for (int i = 0; i < CT; i++) {
        float v = alpha[i] + tr[i * CT + j];
        if (v > best) { best = v; bi = i; }
      }
      bp[b][s - 1][j] = (unsigned char)bi;
      na[j] = best + eb[s * CT + j];
    }
    for (int j = 0; j < CT; j++) alpha[j] = na[j];
  }
  float best = -3.0e38f; int last = 0;
  for (int j = 0; j < CT; j++) {
    float v = alpha[j] + en[j];
    if (v > best) { best = v; last = j; }
  }
  out[1 + b * CS + (CS - 1)] = (float)last;
  int tag = last;
  for (int s = CS - 2; s >= 0; s--) {
    tag = bp[b][s][tag];
    out[1 + b * CS + s] = (float)tag;
  }
}

// ---------------- host-side orchestration ----------------
extern "C" void kernel_launch(void* const* d_in, const int* in_sizes, int n_in,
                              void* d_out, int out_size, void* d_ws, size_t ws_size,
                              hipStream_t stream) {
  (void)in_sizes; (void)n_in; (void)out_size; (void)ws_size;
  const float* word_emb = (const float*)d_in[0];
  const float* pos_emb  = (const float*)d_in[1];
  const float* type_emb = (const float*)d_in[2];
  const float* emb_g    = (const float*)d_in[3];
  const float* emb_b    = (const float*)d_in[4];
  const float* Wq   = (const float*)d_in[5];
  const float* Wk   = (const float*)d_in[6];
  const float* Wv   = (const float*)d_in[7];
  const float* Wo   = (const float*)d_in[8];
  const float* Wff1 = (const float*)d_in[9];
  const float* Wff2 = (const float*)d_in[10];
  const float* bq   = (const float*)d_in[11];
  const float* bk   = (const float*)d_in[12];
  const float* bv   = (const float*)d_in[13];
  const float* bo   = (const float*)d_in[14];
  const float* bff1 = (const float*)d_in[15];
  const float* bff2 = (const float*)d_in[16];
  const float* ln1g = (const float*)d_in[17];
  const float* ln2g = (const float*)d_in[18];
  const float* ln1b = (const float*)d_in[19];
  const float* ln2b = (const float*)d_in[20];
  const float* Wih_f = (const float*)d_in[21];
  const float* Whh_f = (const float*)d_in[22];
  const float* bih_f = (const float*)d_in[23];
  const float* bhh_f = (const float*)d_in[24];
  const float* Wih_b = (const float*)d_in[25];
  const float* Whh_b = (const float*)d_in[26];
  const float* bih_b = (const float*)d_in[27];
  const float* bhh_b = (const float*)d_in[28];
  const float* out_W = (const float*)d_in[29];
  const float* out_b = (const float*)d_in[30];
  const float* crf_start = (const float*)d_in[31];
  const float* crf_end   = (const float*)d_in[32];
  const float* crf_trans = (const float*)d_in[33];
  const int* input_ids = (const int*)d_in[34];
  const int* attn_mask = (const int*)d_in[35];
  const int* tags      = (const int*)d_in[36];
  float* out = (float*)d_out;

  // workspace carve (256B aligned)
  char* w = (char*)d_ws;
  auto alloc = [&](size_t bytes) { void* p = (void*)w; w += (bytes + 255) & ~(size_t)255; return p; };
  float*  x    = (float*) alloc((size_t)CM * CH * 4);
  __bf16* xb   = (__bf16*)alloc((size_t)CM * CH * 2);
  __bf16* qb   = (__bf16*)alloc((size_t)CM * CH * 2);
  __bf16* kb   = (__bf16*)alloc((size_t)CM * CH * 2);
  __bf16* vb   = (__bf16*)alloc((size_t)CM * CH * 2);
  float*  sc   = (float*) alloc((size_t)CB * CNH * CS * CS * 4);
  size_t  prBytes = (size_t)CB * CNH * SPAD * KPAD * 2;
  __bf16* pr   = (__bf16*)alloc(prBytes);
  __bf16* ctx  = (__bf16*)alloc((size_t)CM * CH * 2);
  float*  tmp  = (float*) alloc((size_t)CM * CH * 4);
  __bf16* ffb  = (__bf16*)alloc((size_t)CM * CFF * 2);
  float*  wihT = (float*) alloc((size_t)2 * CH * 4 * CLH * 4);
  __bf16* whhT = (__bf16*)alloc((size_t)2 * 4 * CLH * CLH * 2);
  float*  bc   = (float*) alloc((size_t)2 * 4 * CLH * 4);
  float*  xp   = (float*) alloc((size_t)2 * CM * 4 * CLH * 4);
  __bf16* lo   = (__bf16*)alloc((size_t)CM * CHID * 2);
  float*  em   = (float*) alloc((size_t)CM * CT * 4);

  hipMemsetAsync(pr, 0, prBytes, stream);

  embed_ln_kernel<<<CM, 256, 0, stream>>>(input_ids, word_emb, pos_emb, type_emb,
                                          emb_g, emb_b, x, xb);
  for (int l = 0; l < CL; l++) {
    const float* Wq_l = Wq + (size_t)l * CH * CH;
    const float* Wk_l = Wk + (size_t)l * CH * CH;
    const float* Wv_l = Wv + (size_t)l * CH * CH;
    const float* Wo_l = Wo + (size_t)l * CH * CH;
    const float* W1_l = Wff1 + (size_t)l * CH * CFF;
    const float* W2_l = Wff2 + (size_t)l * CFF * CH;
    dim3 g6(6, GY), g24(24, GY);
    gemm_kernel<<<g6, 256, 0, stream>>>(xb, CH, Wq_l, CH, bq + l * CH, nullptr, qb, CH, CM, CH, CH, 0);
    gemm_kernel<<<g6, 256, 0, stream>>>(xb, CH, Wk_l, CH, bk + l * CH, nullptr, kb, CH, CM, CH, CH, 0);
    gemm_kernel<<<g6, 256, 0, stream>>>(xb, CH, Wv_l, CH, bv + l * CH, nullptr, vb, CH, CM, CH, CH, 0);
    scores_kernel<<<CB * CNH, 256, 0, stream>>>(qb, kb, attn_mask, sc);
    softmax_kernel<<<(CB * CNH * CS) / 4, 128, 0, stream>>>(sc, pr);
    ctx_kernel<<<CB * CNH, 256, 0, stream>>>(pr, vb, ctx);
    gemm_kernel<<<g6, 256, 0, stream>>>(ctx, CH, Wo_l, CH, bo + l * CH, tmp, nullptr, CH, CM, CH, CH, 0);
    addln_kernel<<<CM, 256, 0, stream>>>(x, tmp, ln1g + l * CH, ln1b + l * CH, x, xb);
    gemm_kernel<<<g24, 256, 0, stream>>>(xb, CH, W1_l, CFF, bff1 + l * CFF, nullptr, ffb, CFF, CM, CFF, CH, 1);
    gemm_kernel<<<g6, 256, 0, stream>>>(ffb, CFF, W2_l, CH, bff2 + l * CH, tmp, nullptr, CH, CM, CH, CFF, 0);
    addln_kernel<<<CM, 256, 0, stream>>>(x, tmp, ln2g + l * CH, ln2b + l * CH, x, xb);
  }

  int nW = 4 * CLH * CH;
  transpose_f32_kernel<<<(nW + 255) / 256, 256, 0, stream>>>(Wih_f, wihT, 4 * CLH, CH);
  transpose_f32_kernel<<<(nW + 255) / 256, 256, 0, stream>>>(Wih_b, wihT + (size_t)CH * 4 * CLH, 4 * CLH, CH);
  int nH = 4 * CLH * CLH;
  cvt_bf16_kernel<<<(nH + 255) / 256, 256, 0, stream>>>(Whh_f, whhT, nH);
  cvt_bf16_kernel<<<(nH + 255) / 256, 256, 0, stream>>>(Whh_b, whhT + nH, nH);
  addvec_kernel<<<2, 256, 0, stream>>>(bih_f, bhh_f, bc, 4 * CLH);
  addvec_kernel<<<2, 256, 0, stream>>>(bih_b, bhh_b, bc + 4 * CLH, 4 * CLH);

  dim3 g4(4, GY);
  gemm_kernel<<<g4, 256, 0, stream>>>(xb, CH, wihT, 4 * CLH, bc, xp, nullptr, 4 * CLH,
                                      CM, 4 * CLH, CH, 0);
  gemm_kernel<<<g4, 256, 0, stream>>>(xb, CH, wihT + (size_t)CH * 4 * CLH, 4 * CLH,
                                      bc + 4 * CLH, xp + (size_t)CM * 4 * CLH, nullptr, 4 * CLH,
                                      CM, 4 * CLH, CH, 0);
  lstm_kernel<<<2, 512, 0, stream>>>(xp, whhT, lo);

  dim3 g1(1, GY);
  gemm_kernel<<<g1, 256, 0, stream>>>(lo, CHID, out_W, CT, out_b, em, nullptr, CT,
                                      CM, CT, CHID, 0);
  crf_nll_kernel<<<1, 32, 0, stream>>>(em, tags, crf_start, crf_end, crf_trans, out);
  viterbi_kernel<<<1, 32, 0, stream>>>(em, crf_start, crf_end, crf_trans, out);
}
